// SimpleGenerator_29351806501196
// MI455X (gfx1250) — compile-verified, matrix operand order checked by reading, not run on hardware
//
#include <hip/hip_runtime.h>
#include <stdint.h>

// Problem sizes
#define Bm   64
#define Tm   256
#define EMB  512
#define Hm   1024
#define G3   3072      // 3*H
#define KV   1536      // EMB + H
#define COLS 16        // H columns owned per workgroup in the scan
#define NWG  64        // H / COLS persistent workgroups
#define KX   16        // EMB/32 wmma k-steps (x part)
#define KH   32        // H/32 wmma k-steps (h part)
#define KHH  16        // KH/2: k-steps per khalf wave

typedef __attribute__((ext_vector_type(16))) __bf16 v16bf;
typedef __attribute__((ext_vector_type(8)))  __bf16 v8bf;
typedef __attribute__((ext_vector_type(8)))  float  v8f;
typedef __attribute__((ext_vector_type(4)))  unsigned int tdm_u4;
typedef __attribute__((ext_vector_type(8)))  int          tdm_i8;
typedef __attribute__((ext_vector_type(4)))  int          tdm_i4;

union ABf { v16bf v; v8bf h[2]; __bf16 e[16]; };
union F8  { v8f v; float4 f4[2]; float e[8]; };

// Workspace layout (bytes)
#define WCAT_OFF 0u          // [3072][1536] bf16  = 9,437,184
#define FC1_OFF  9437184u    // [512][1024] bf16   = 1,048,576
#define XBF_OFF  10485760u   // [64*256][512] bf16 = 16,777,216
#define HBF_OFF  27262976u   // 2 x [64][1024] bf16 = 262,144 (ping-pong)
#define BAR_OFF  27525120u   // barrier state
#define GI_OFF   27525248u   // [T][3H][64] f32 = 201,326,592 (optional)
#define GI_BYTES 201326592u

__device__ __forceinline__ v8f wmma_bf16(v16bf a, v16bf b, v8f c) {
  return __builtin_amdgcn_wmma_f32_16x16x32_bf16(false, a, false, b,
                                                 (short)0, c, false, false);
}

// ---------------- convert / init kernels (fully parallel) ----------------

__global__ __launch_bounds__(256) void k_convert_wcat(
    const float* __restrict__ Wih, const float* __restrict__ Whh,
    __bf16* __restrict__ wcat) {
  int idx = blockIdx.x * blockDim.x + threadIdx.x;
  const int total = G3 * KV;
  for (; idx < total; idx += gridDim.x * blockDim.x) {
    int n = idx / KV, k = idx - n * KV;
    float v = (k < EMB) ? Wih[n * EMB + k] : Whh[n * Hm + (k - EMB)];
    wcat[idx] = (__bf16)v;
  }
}

__global__ __launch_bounds__(256) void k_convert_lin(
    const float* __restrict__ src, __bf16* __restrict__ dst, int n) {
  int idx = blockIdx.x * blockDim.x + threadIdx.x;
  for (; idx < n; idx += gridDim.x * blockDim.x)
    dst[idx] = (__bf16)src[idx];
}

__global__ __launch_bounds__(256) void k_init_h0(
    const float* __restrict__ y, const float* __restrict__ z,
    __bf16* __restrict__ hbf0, unsigned* __restrict__ bar) {
  int idx = blockIdx.x * blockDim.x + threadIdx.x;
  if (idx == 0) { bar[0] = 0u; bar[1] = 0u; }
  const int total = Bm * Hm;
  for (; idx < total; idx += gridDim.x * blockDim.x) {
    int b = idx >> 10, c = idx & 1023;
    float v = (c < 256) ? y[b * 256 + c] : z[b * 768 + (c - 256)];
    hbf0[idx] = (__bf16)v;
  }
}

// ---------------- gi = x @ W_ih^T, tile-major output ----------------
// gi[t][gcol][m] so the scan can init accumulators with contiguous v8f loads.

__global__ __launch_bounds__(256) void k_gi_gemm(
    const __bf16* __restrict__ xbf,   // [B*T][EMB]
    const __bf16* __restrict__ wcat,  // [3H][KV], x part is k < EMB
    float* __restrict__ gi) {         // [T][G3][64]
  const int lane = threadIdx.x & 31;
  const int wv   = threadIdx.x >> 5;
  const int tile = blockIdx.x * 8 + wv;       // T * 4 * 192 tiles
  const int t    = tile / 768;
  const int rem  = tile - t * 768;
  const int mt   = rem / 192;
  const int nt   = rem - mt * 192;
  const int rowin = lane & 15;
  const int hiK  = (lane & 16) ? 8 : 0;
  const int b    = mt * 16 + rowin;
  const int n    = nt * 16 + rowin;

  const __bf16* ax = xbf + ((size_t)b * Tm + t) * EMB;
  const __bf16* bw = wcat + (size_t)n * KV;

  v8f acc = {};
  for (int ks = 0; ks < KX; ++ks) {
    int kb = ks * 32 + hiK;
    ABf a, w;
    a.h[0] = *(const v8bf*)(ax + kb);
    a.h[1] = *(const v8bf*)(ax + kb + 16);
    w.h[0] = *(const v8bf*)(bw + kb);
    w.h[1] = *(const v8bf*)(bw + kb + 16);
    acc = wmma_bf16(a.v, w.v, acc);
  }
  float* dst = gi + (((size_t)t * G3 + n) * 64 + mt * 16 + hiK);
  F8 o; o.v = acc;
  *(float4*)dst = o.f4[0];
  *(float4*)(dst + 4) = o.f4[1];
}

// ---------------- persistent GRU scan kernel ----------------
// 64 WGs x 256 threads (8 wave32s = (mt 0..3, khalf 0..1)). WG owns H-columns
// [wg*16, wg*16+16) for all 3 gates. Wave (mt,0) owns the output tile: batch
// rows [16mt,16mt+16), h tile in registers for the whole scan, gi accumulator
// init and gate math. Wave (mt,1) computes the second K-half partials and
// ships them through LDS. h_t is staged into LDS by the Tensor Data Mover.

__global__ __launch_bounds__(256) void k_gru_scan(
    const __bf16* __restrict__ wcat,   // [3H][KV] bf16
    const __bf16* __restrict__ xbf,    // [B*T][EMB] bf16 (fallback path)
    __bf16* __restrict__ hbf,          // 2 x [B][H] bf16 ping-pong
    const float* __restrict__ b_ih, const float* __restrict__ b_hh,
    const float* __restrict__ gi, int use_gi,
    float* __restrict__ hidden,        // [B][T][H] f32 (section of d_out)
    unsigned* __restrict__ bar) {
  __shared__ __bf16 ldsW[3 * KH * 32 * 16];  //  98,304 B: W_hh B-fragments
  __shared__ __bf16 ldsH[Bm * Hm];           // 131,072 B: h_t row-major (TDM)
  __shared__ float  ldsP[4 * 32 * 24];       //  12,288 B: khalf=1 partials

  const int tid = threadIdx.x;
  const int c0  = blockIdx.x * COLS;

  // one-time stage of W_hh slice into wmma B-fragment lane order
  for (int fl = tid; fl < 3 * KH * 32; fl += blockDim.x) {
    int l    = fl & 31;
    int frag = fl >> 5;                 // gate*KH + ksh
    int gate = frag / KH;
    int ksh  = frag - gate * KH;
    int n    = gate * Hm + c0 + (l & 15);
    int kb   = (KX + ksh) * 32 + ((l & 16) ? 8 : 0);
    const uint4* s0 = (const uint4*)(wcat + (size_t)n * KV + kb);
    const uint4* s1 = (const uint4*)(wcat + (size_t)n * KV + kb + 16);
    uint4* d = (uint4*)(&ldsW[fl * 16]);
    d[0] = s0[0];
    d[1] = s1[0];
  }
  __syncthreads();

  const int lane  = tid & 31;
  const int wv    = tid >> 5;
  const int mt    = wv & 3;
  const int khalf = wv >> 2;                // 0: ks 0..15, 1: ks 16..31
  const int rowin = lane & 15;
  const int hiK   = (lane & 16) ? 8 : 0;
  const int b0    = mt * 16;
  const int arow  = b0 + rowin;             // A row (batch index)
  const int colg  = c0 + rowin;             // C column in [0,H)
  const int ksb   = khalf * KHH;            // this wave's h k-step base

  const float br  = b_ih[colg]          + b_hh[colg];
  const float bz  = b_ih[Hm + colg]     + b_hh[Hm + colg];
  const float bin = b_ih[2 * Hm + colg];
  const float bhn = b_hh[2 * Hm + colg];

  float hold[8];
#pragma unroll
  for (int i = 0; i < 8; ++i)
    hold[i] = (float)hbf[(size_t)(b0 + i + hiK) * Hm + colg];

  const __bf16* xrow = xbf + (size_t)arow * Tm * EMB;
  float* myP = &ldsP[(mt * 32 + lane) * 24];

  for (int t = 0; t < Tm; ++t) {
    const int bufR = t & 1;
    const __bf16* hR = hbf + (size_t)bufR * Bm * Hm;
    __bf16* hW = hbf + (size_t)(1 - bufR) * Bm * Hm;

    // Stage h_t [64][1024] bf16 into LDS via the Tensor Data Mover.
    if (wv == 0) {
      uint64_t gaddr = (uint64_t)(uintptr_t)hR;
      uint32_t ldso  = (uint32_t)(uintptr_t)(&ldsH[0]);
      tdm_u4 g0;
      g0.x = 1u;                                   // count=1 user descriptor
      g0.y = ldso;                                 // lds_addr
      g0.z = (uint32_t)gaddr;                      // global_addr[31:0]
      g0.w = (uint32_t)((gaddr >> 32) & 0x01FFFFFFu) | (2u << 30); // type=2
      tdm_i8 g1;
      g1[0] = 0x00010000;            // mask=0, data_size=1 (2 bytes)
      g1[1] = (int)(1024u << 16);    // tensor_dim0 = 1024 (low 16)
      g1[2] = (int)(64u << 16);      // tensor_dim1 = 64 (low 16)
      g1[3] = (int)(1024u << 16);    // tile_dim0 = 1024
      g1[4] = 64;                    // tile_dim1 = 64, tile_dim2 = 0
      g1[5] = 1024;                  // tensor_dim0_stride = 1024
      g1[6] = 0;
      g1[7] = 0;
      tdm_i4 gz = {};
#if __clang_major__ >= 23
      tdm_i8 gz8 = {};
      __builtin_amdgcn_tensor_load_to_lds(g0, g1, gz, gz, gz8, 0);
#else
      __builtin_amdgcn_tensor_load_to_lds(g0, g1, gz, gz, 0);
#endif
      __builtin_amdgcn_s_wait_tensorcnt(0);
    }
    __syncthreads();

    v8f accR = {}, accZ = {}, accIN = {}, accHN = {};
    if (khalf == 0) {
      if (use_gi) {
        const float* gbase = gi + ((size_t)t * G3) * 64;
        const float* pr = gbase + ((size_t)(0 * Hm + colg)) * 64 + b0 + hiK;
        const float* pz = gbase + ((size_t)(1 * Hm + colg)) * 64 + b0 + hiK;
        const float* pn = gbase + ((size_t)(2 * Hm + colg)) * 64 + b0 + hiK;
        F8 r0, z0, n0;
        r0.f4[0] = *(const float4*)pr; r0.f4[1] = *(const float4*)(pr + 4);
        z0.f4[0] = *(const float4*)pz; z0.f4[1] = *(const float4*)(pz + 4);
        n0.f4[0] = *(const float4*)pn; n0.f4[1] = *(const float4*)(pn + 4);
        accR = r0.v; accZ = z0.v; accIN = n0.v;
        if (t + 1 < Tm) {  // prefetch next step's gate tiles into GL2
          const float* gnext = gi + ((size_t)(t + 1) * G3) * 64;
          __builtin_prefetch(gnext + ((size_t)colg) * 64 + b0, 0, 1);
          __builtin_prefetch(gnext + ((size_t)(Hm + colg)) * 64 + b0, 0, 1);
          __builtin_prefetch(gnext + ((size_t)(2 * Hm + colg)) * 64 + b0, 0, 1);
        }
      } else {
        // fused x part: A from global x, B gathered from global W (L2-hot)
        for (int ks = 0; ks < KX; ++ks) {
          int kb = ks * 32 + hiK;
          ABf a, bR, bZ, bN;
          a.h[0]  = *(const v8bf*)(xrow + (size_t)t * EMB + kb);
          a.h[1]  = *(const v8bf*)(xrow + (size_t)t * EMB + kb + 16);
          bR.h[0] = *(const v8bf*)(wcat + (size_t)colg * KV + kb);
          bR.h[1] = *(const v8bf*)(wcat + (size_t)colg * KV + kb + 16);
          bZ.h[0] = *(const v8bf*)(wcat + (size_t)(Hm + colg) * KV + kb);
          bZ.h[1] = *(const v8bf*)(wcat + (size_t)(Hm + colg) * KV + kb + 16);
          bN.h[0] = *(const v8bf*)(wcat + (size_t)(2 * Hm + colg) * KV + kb);
          bN.h[1] = *(const v8bf*)(wcat + (size_t)(2 * Hm + colg) * KV + kb + 16);
          accR  = wmma_bf16(a.v, bR.v, accR);
          accZ  = wmma_bf16(a.v, bZ.v, accZ);
          accIN = wmma_bf16(a.v, bN.v, accIN);
        }
      }
    }

    // h part: each khalf covers 16 of 32 k-steps. A from TDM-staged LDS h_t,
    // B from LDS W fragments.
    for (int ks = 0; ks < KHH; ++ks) {
      int kabs = ksb + ks;
      int kb = kabs * 32 + hiK;
      ABf a;
      a.h[0] = *(const v8bf*)(&ldsH[arow * Hm + kb]);
      a.h[1] = *(const v8bf*)(&ldsH[arow * Hm + kb + 16]);
      v16bf bR = *(const v16bf*)(&ldsW[((0 * KH + kabs) * 32 + lane) * 16]);
      v16bf bZ = *(const v16bf*)(&ldsW[((1 * KH + kabs) * 32 + lane) * 16]);
      v16bf bN = *(const v16bf*)(&ldsW[((2 * KH + kabs) * 32 + lane) * 16]);
      accR  = wmma_bf16(a.v, bR, accR);
      accZ  = wmma_bf16(a.v, bZ, accZ);
      accHN = wmma_bf16(a.v, bN, accHN);
    }

    // khalf=1 ships partial tiles through LDS
    if (khalf == 1) {
      F8 r, z, h;
      r.v = accR; z.v = accZ; h.v = accHN;
      *(float4*)(myP + 0)  = r.f4[0]; *(float4*)(myP + 4)  = r.f4[1];
      *(float4*)(myP + 8)  = z.f4[0]; *(float4*)(myP + 12) = z.f4[1];
      *(float4*)(myP + 16) = h.f4[0]; *(float4*)(myP + 20) = h.f4[1];
    }
    __syncthreads();

    if (khalf == 0) {
      F8 r1, z1, h1;
      r1.f4[0] = *(const float4*)(myP + 0);  r1.f4[1] = *(const float4*)(myP + 4);
      z1.f4[0] = *(const float4*)(myP + 8);  z1.f4[1] = *(const float4*)(myP + 12);
      h1.f4[0] = *(const float4*)(myP + 16); h1.f4[1] = *(const float4*)(myP + 20);
      accR  = accR  + r1.v;
      accZ  = accZ  + z1.v;
      accHN = accHN + h1.v;

      // gates + state update + stores
#pragma unroll
      for (int i = 0; i < 8; ++i) {
        int brow = b0 + i + hiK;
        float r  = 1.f / (1.f + __expf(-(accR[i] + br)));
        float zg = 1.f / (1.f + __expf(-(accZ[i] + bz)));
        float nn = tanhf(accIN[i] + bin + r * (accHN[i] + bhn));
        float hn = (1.f - zg) * nn + zg * hold[i];
        hold[i] = hn;
        hidden[((size_t)brow * Tm + t) * Hm + colg] = hn;
        hW[(size_t)brow * Hm + colg] = (__bf16)hn;
      }
    }

    // grid-wide barrier (monotonic count; NWG resident WGs)
    __syncthreads();
    if (tid == 0) {
      __threadfence();
      unsigned arrived = atomicAdd(&bar[0], 1u) + 1u;
      unsigned target  = (unsigned)gridDim.x * (unsigned)(t + 1);
      if (arrived == target) atomicExch(&bar[1], (unsigned)(t + 1));
      volatile unsigned* vs = &bar[1];
      while (*vs < (unsigned)(t + 1)) __builtin_amdgcn_s_sleep(1);
      __threadfence();
    }
    __syncthreads();
  }
}

// ---------------- output projection: outputs = hidden @ fc1_w^T + b ----------------

__global__ __launch_bounds__(256) void k_out_gemm(
    const float* __restrict__ hidden,   // [B*T][H] f32
    const __bf16* __restrict__ fc1bf,   // [EMB][H] bf16
    const float* __restrict__ fc1_b,
    float* __restrict__ out) {          // [B*T][EMB] f32
  const int lane = threadIdx.x & 31;
  const int wv   = threadIdx.x >> 5;
  const int tile = blockIdx.x * 8 + wv;
  const int mt   = tile >> 5;
  const int nt   = tile & 31;
  const int rowin = lane & 15;
  const int hiK  = (lane & 16) ? 8 : 0;
  const int arow = mt * 16 + rowin;
  const int ncol = nt * 16 + rowin;

  const float*  ha = hidden + (size_t)arow * Hm;
  const __bf16* bb = fc1bf + (size_t)ncol * Hm;

  v8f acc = {};
  for (int ks = 0; ks < Hm / 32; ++ks) {
    int kb = ks * 32 + hiK;
    float4 f0 = *(const float4*)(ha + kb);
    float4 f1 = *(const float4*)(ha + kb + 4);
    float4 f2 = *(const float4*)(ha + kb + 16);
    float4 f3 = *(const float4*)(ha + kb + 20);
    ABf a;
    a.e[0]  = (__bf16)f0.x; a.e[1]  = (__bf16)f0.y;
    a.e[2]  = (__bf16)f0.z; a.e[3]  = (__bf16)f0.w;
    a.e[4]  = (__bf16)f1.x; a.e[5]  = (__bf16)f1.y;
    a.e[6]  = (__bf16)f1.z; a.e[7]  = (__bf16)f1.w;
    a.e[8]  = (__bf16)f2.x; a.e[9]  = (__bf16)f2.y;
    a.e[10] = (__bf16)f2.z; a.e[11] = (__bf16)f2.w;
    a.e[12] = (__bf16)f3.x; a.e[13] = (__bf16)f3.y;
    a.e[14] = (__bf16)f3.z; a.e[15] = (__bf16)f3.w;
    ABf b;
    b.h[0] = *(const v8bf*)(bb + kb);
    b.h[1] = *(const v8bf*)(bb + kb + 16);
    acc = wmma_bf16(a.v, b.v, acc);
  }
  const float bias = fc1_b[ncol];
#pragma unroll
  for (int i = 0; i < 8; ++i) {
    int r = mt * 16 + i + hiK;
    out[(size_t)r * EMB + ncol] = acc[i] + bias;
  }
}

// ---------------- host-side launch ----------------

extern "C" void kernel_launch(void* const* d_in, const int* in_sizes, int n_in,
                              void* d_out, int out_size, void* d_ws, size_t ws_size,
                              hipStream_t stream) {
  const float* y    = (const float*)d_in[0];
  const float* z    = (const float*)d_in[1];
  const float* x    = (const float*)d_in[2];
  const float* Wih  = (const float*)d_in[3];
  const float* Whh  = (const float*)d_in[4];
  const float* bih  = (const float*)d_in[5];
  const float* bhh  = (const float*)d_in[6];
  const float* fc1w = (const float*)d_in[7];
  const float* fc1b = (const float*)d_in[8];

  float* out    = (float*)d_out;                 // [B,T,EMB]
  float* hidden = out + (size_t)Bm * Tm * EMB;   // [B,T,H]

  char* ws = (char*)d_ws;
  __bf16*   wcat  = (__bf16*)(ws + WCAT_OFF);
  __bf16*   fc1bf = (__bf16*)(ws + FC1_OFF);
  __bf16*   xbf   = (__bf16*)(ws + XBF_OFF);
  __bf16*   hbf   = (__bf16*)(ws + HBF_OFF);
  unsigned* bar   = (unsigned*)(ws + BAR_OFF);
  float*    gip   = (float*)(ws + GI_OFF);

  const int use_gi = (ws_size >= (size_t)GI_OFF + (size_t)GI_BYTES) ? 1 : 0;

  k_convert_wcat<<<512, 256, 0, stream>>>(Wih, Whh, wcat);
  k_convert_lin<<<256, 256, 0, stream>>>(fc1w, fc1bf, EMB * Hm);
  k_convert_lin<<<1024, 256, 0, stream>>>(x, xbf, Bm * Tm * EMB);
  k_init_h0<<<64, 256, 0, stream>>>(y, z, hbf, bar);

  if (use_gi) {
    // T * 4 * 192 tiles / 8 waves per block
    k_gi_gemm<<<Tm * 4 * 192 / 8, 256, 0, stream>>>(xbf, wcat, gip);
  }

  k_gru_scan<<<NWG, 256, 0, stream>>>(wcat, xbf, hbf, bih, bhh, gip, use_gi,
                                      hidden, bar);

  k_out_gemm<<<(Bm * Tm / 16) * (EMB / 16) / 8, 256, 0, stream>>>(
      hidden, fc1bf, fc1b, out);
}
